// RBFLayer_16320875725288
// MI455X (gfx1250) — compile-verified, matrix operand order checked by reading, not run on hardware
//
#include <hip/hip_runtime.h>
#include <hip/hip_bf16.h>

// RBF layer: out[b,o] = ||x_b||^2 - 2 (x@W)[b,o] + ||w_o||^2
// Compute-bound GEMM (68.7 GFLOP vs 176 MB traffic). bf16x3-split WMMA
// (hi*hi + hi*lo + lo*hi) on v_wmma_f32_16x16x32_bf16: ~2.7x the MAC rate of
// the native fp32 WMMA (16x16x4) at fp32-class accuracy.
// v2: double-buffered LDS (one barrier/iter), register prefetch of next K-tile,
//     4x4-block B transpose staging (b64 stores), global_prefetch of K+2 tile.

typedef __attribute__((ext_vector_type(16))) __bf16 v16bf;
typedef __attribute__((ext_vector_type(8)))  __bf16 v8bf;
typedef __attribute__((ext_vector_type(4)))  __bf16 v4bf;
typedef __attribute__((ext_vector_type(8)))  float  v8f;

#define IN_DIM  1024
#define OUT_DIM 4096
#define BM 128
#define BN 128
#define BK 32
#define LDSK 40   // padded k-stride (elements): 80B row pitch -> conflict-free b128 reads

union FragBF { v16bf v; v8bf h[2]; };

struct HiLo { __bf16 hi; __bf16 lo; };

static __device__ inline HiLo split_bf16(float f) {
  HiLo r;
  r.hi = (__bf16)f;
  r.lo = (__bf16)(f - (float)r.hi);
  return r;
}

static __device__ inline v8f wmma_bf16(v16bf a, v16bf b, v8f c) {
  return __builtin_amdgcn_wmma_f32_16x16x32_bf16(
      /*neg_a=*/false, a, /*neg_b=*/false, b,
      /*c_mod=*/(short)0, c, /*reuse_a=*/false, /*reuse_b=*/false);
}

// ---------------- row norms: x_sq[b] = sum_i x[b,i]^2 ----------------
__global__ __launch_bounds__(256) void xsq_kernel(const float* __restrict__ x,
                                                  float* __restrict__ xsq, int B) {
  const int wave = threadIdx.x >> 5;
  const int lane = threadIdx.x & 31;
  const int row  = blockIdx.x * 8 + wave;
  if (row >= B) return;
  const float* p = x + (size_t)row * IN_DIM;
  float s = 0.f;
  #pragma unroll 4
  for (int j = lane; j < IN_DIM; j += 32) { float v = p[j]; s += v * v; }
  #pragma unroll
  for (int off = 16; off >= 1; off >>= 1) s += __shfl_xor(s, off, 32);
  if (lane == 0) xsq[row] = s;
}

// ---------------- col norms: w_sq[o] = sum_i W[i,o]^2 ----------------
__global__ __launch_bounds__(256) void wsq_kernel(const float* __restrict__ w,
                                                  float* __restrict__ wsq) {
  const int o = blockIdx.x * blockDim.x + threadIdx.x;
  float s = 0.f;
  #pragma unroll 4
  for (int i = 0; i < IN_DIM; ++i) { float v = w[(size_t)i * OUT_DIM + o]; s += v * v; }
  wsq[o] = s;
}

// ---------------- main tiled WMMA GEMM + epilogue ----------------
__global__ __launch_bounds__(256) void rbf_gemm(const float* __restrict__ x,
                                                const float* __restrict__ w,
                                                const float* __restrict__ xsq,
                                                const float* __restrict__ wsq,
                                                float* __restrict__ out) {
  // double-buffered staging: 2 * (A hi/lo + B hi/lo) = 80 KB (WGP has 320 KB)
  __shared__ __align__(16) __bf16 AhS[2][BM * LDSK];
  __shared__ __align__(16) __bf16 AlS[2][BM * LDSK];
  __shared__ __align__(16) __bf16 BhS[2][BN * LDSK];
  __shared__ __align__(16) __bf16 BlS[2][BN * LDSK];

  const int t     = threadIdx.x;
  const int wave  = t >> 5;
  const int lane  = t & 31;
  const int wm    = wave & 3;            // 4 waves along M
  const int wn    = wave >> 2;           // 2 waves along N
  const int m_off = wm * 32;
  const int n_off = wn * 64;
  const int Mb    = blockIdx.y * BM;
  const int Nb    = blockIdx.x * BN;
  const int lhalf = lane >> 4;           // 0: lanes 0-15, 1: lanes 16-31
  const int l16   = lane & 15;

  v8f acc[2][4];
  #pragma unroll
  for (int mi = 0; mi < 2; ++mi)
    #pragma unroll
    for (int ni = 0; ni < 4; ++ni)
      acc[mi][ni] = (v8f){0.f, 0.f, 0.f, 0.f, 0.f, 0.f, 0.f, 0.f};

  // staging thread mapping
  const int a_r4 = t >> 3;          // 0..31  A: row within pass (4 passes of 32 rows)
  const int a_k4 = (t & 7) * 4;     // 0..28  A: k (float4)
  const int b_n4 = (t & 31) * 4;    // 0..124 B: n base of 4x4 block
  const int b_k4 = (t >> 5) * 4;    // 0..28  B: k base of 4x4 block

  // ---- register prefetch of first K-tile ----
  float4 ax[4], bw[4];
  #pragma unroll
  for (int p = 0; p < 4; ++p) {
    ax[p] = *(const float4*)(x + (size_t)(Mb + p * 32 + a_r4) * IN_DIM + a_k4);
    bw[p] = *(const float4*)(w + (size_t)(b_k4 + p) * OUT_DIM + Nb + b_n4);
  }

  int cur = 0;
  for (int k0 = 0; k0 < IN_DIM; k0 += BK, cur ^= 1) {
    // ---- stage A tile from regs (BM x BK fp32 -> bf16 hi/lo), layout [m][k] ----
    #pragma unroll
    for (int p = 0; p < 4; ++p) {
      const int row = p * 32 + a_r4;
      const float4 v = ax[p];
      v4bf h, l;
      HiLo s0 = split_bf16(v.x); h[0] = s0.hi; l[0] = s0.lo;
      HiLo s1 = split_bf16(v.y); h[1] = s1.hi; l[1] = s1.lo;
      HiLo s2 = split_bf16(v.z); h[2] = s2.hi; l[2] = s2.lo;
      HiLo s3 = split_bf16(v.w); h[3] = s3.hi; l[3] = s3.lo;
      *(v4bf*)&AhS[cur][row * LDSK + a_k4] = h;
      *(v4bf*)&AlS[cur][row * LDSK + a_k4] = l;
    }
    // ---- stage B 4x4 block from regs, transposed to [n][k] (k contiguous) ----
    #pragma unroll
    for (int j = 0; j < 4; ++j) {       // j = n offset within block
      v4bf h, l;
      #pragma unroll
      for (int i = 0; i < 4; ++i) {     // i = k offset within block
        const float f = (j == 0) ? bw[i].x : (j == 1) ? bw[i].y : (j == 2) ? bw[i].z : bw[i].w;
        HiLo s = split_bf16(f);
        h[i] = s.hi; l[i] = s.lo;
      }
      *(v4bf*)&BhS[cur][(b_n4 + j) * LDSK + b_k4] = h;
      *(v4bf*)&BlS[cur][(b_n4 + j) * LDSK + b_k4] = l;
    }

    // ---- issue next K-tile global loads (overlap with WMMA phase) ----
    const int kn = k0 + BK;
    if (kn < IN_DIM) {
      #pragma unroll
      for (int p = 0; p < 4; ++p) {
        ax[p] = *(const float4*)(x + (size_t)(Mb + p * 32 + a_r4) * IN_DIM + kn + a_k4);
        bw[p] = *(const float4*)(w + (size_t)(kn + b_k4 + p) * OUT_DIM + Nb + b_n4);
      }
    }
    // ---- prefetch K+2 tile into L2 (global_prefetch_b8) ----
    const int kp = k0 + 2 * BK;
    if (kp < IN_DIM) {
      __builtin_prefetch(x + (size_t)(Mb + a_r4) * IN_DIM + kp + a_k4, 0, 3);
      __builtin_prefetch(w + (size_t)(kp + b_k4) * OUT_DIM + Nb + b_n4, 0, 3);
    }

    __syncthreads();   // staging of `cur` visible; double-buffer -> single barrier

    // ---- A fragments for this wave's 2 M-tiles ----
    // 16-bit A layout: lanes 0-15 hold K{0..7,16..23}, lanes 16-31 K{8..15,24..31}
    FragBF ah[2], al[2];
    #pragma unroll
    for (int mi = 0; mi < 2; ++mi) {
      const int r  = m_off + mi * 16 + l16;
      const int e0 = r * LDSK + lhalf * 8;
      ah[mi].h[0] = *(const v8bf*)&AhS[cur][e0];
      ah[mi].h[1] = *(const v8bf*)&AhS[cur][e0 + 16];
      al[mi].h[0] = *(const v8bf*)&AlS[cur][e0];
      al[mi].h[1] = *(const v8bf*)&AlS[cur][e0 + 16];
    }
    // ---- per N-tile: load B frags, issue 3 WMMAs per (mi,ni) ----
    #pragma unroll
    for (int ni = 0; ni < 4; ++ni) {
      const int c  = n_off + ni * 16 + l16;
      const int e0 = c * LDSK + lhalf * 8;
      FragBF bh, bl;
      bh.h[0] = *(const v8bf*)&BhS[cur][e0];
      bh.h[1] = *(const v8bf*)&BhS[cur][e0 + 16];
      bl.h[0] = *(const v8bf*)&BlS[cur][e0];
      bl.h[1] = *(const v8bf*)&BlS[cur][e0 + 16];
      #pragma unroll
      for (int mi = 0; mi < 2; ++mi) {
        acc[mi][ni] = wmma_bf16(ah[mi].v, bh.v, acc[mi][ni]);  // hi*hi
        acc[mi][ni] = wmma_bf16(ah[mi].v, bl.v, acc[mi][ni]);  // hi*lo
        acc[mi][ni] = wmma_bf16(al[mi].v, bh.v, acc[mi][ni]);  // lo*hi
      }
    }
    // no trailing barrier: next iteration stages the other buffer; all DS reads
    // of `cur` are consumed by WMMAs before this wave reaches the next barrier.
  }

  // ---- epilogue: out = x_sq[row] + w_sq[col] - 2*cross ----
  // C/D layout: VGPR r -> M = r + 8*lhalf, N = l16
  #pragma unroll
  for (int mi = 0; mi < 2; ++mi) {
    const int rowb = Mb + m_off + mi * 16 + 8 * lhalf;
    float xs[8];
    #pragma unroll
    for (int r = 0; r < 8; ++r) xs[r] = xsq[rowb + r];
    #pragma unroll
    for (int ni = 0; ni < 4; ++ni) {
      const int col = Nb + n_off + ni * 16 + l16;
      const float wsv = wsq[col];
      #pragma unroll
      for (int r = 0; r < 8; ++r) {
        out[(size_t)(rowb + r) * OUT_DIM + col] = xs[r] + wsv - 2.0f * acc[mi][ni][r];
      }
    }
  }
}

extern "C" void kernel_launch(void* const* d_in, const int* in_sizes, int n_in,
                              void* d_out, int out_size, void* d_ws, size_t ws_size,
                              hipStream_t stream) {
  const float* x = (const float*)d_in[0];
  const float* w = (const float*)d_in[1];
  float* out = (float*)d_out;
  const int B = in_sizes[0] / IN_DIM;   // 8192

  float* xsq = (float*)d_ws;            // B floats
  float* wsq = xsq + B;                 // OUT_DIM floats

  xsq_kernel<<<B / 8, 256, 0, stream>>>(x, xsq, B);
  wsq_kernel<<<OUT_DIM / 256, 256, 0, stream>>>(w, wsq);

  dim3 grid(OUT_DIM / BN, B / BM);      // (32, 64)
  rbf_gemm<<<grid, 256, 0, stream>>>(x, w, xsq, wsq, out);
}
